// CIF_Model_19275813224888
// MI455X (gfx1250) — matrix operand order repaired and checked by reading, not compiled, not run on hardware
//
#include <hip/hip_runtime.h>

// CIF model for MI455X (gfx1250, wave32, WMMA).
// Kernel 1: fused depthwise-conv(K=3) + residual + bias + ReLU + GEMV(lin_w)
//           + sigmoid -> alphas[B,T]; H-reduction via v_wmma_f32_16x16x32_f16.
// Kernel 2: sequential integrate-and-fire scan fused with the gather step.
//           alphas staged in LDS (critical scalar chain off VMEM latency),
//           hidden rows prefetched ahead (global_prefetch_b8).

typedef __attribute__((ext_vector_type(16))) _Float16 v16h;
typedef __attribute__((ext_vector_type(8)))  float    v8f;

#define CIF_B 16
#define CIF_T 2048
#define CIF_H 1024
#define WAVES_PER_BLOCK 8   // 256 threads = 8 wave32
#define PREFETCH_AHEAD 8    // rows of hidden prefetched ahead in the scan

// ---------------------------------------------------------------------------
// Kernel 1: alphas.  One wave handles a 16-timestep tile of one batch row.
// Tiles never cross batch boundaries (T % 16 == 0).
// ---------------------------------------------------------------------------
__global__ __launch_bounds__(256) void cif_alphas_kernel(
    const float* __restrict__ hidden,   // [B,T,H]
    const float* __restrict__ conv_w,   // [H,1,3]
    const float* __restrict__ conv_b,   // [H]
    const float* __restrict__ lin_w,    // [1,H]
    const float* __restrict__ lin_b,    // [1]
    float* __restrict__ alphas)         // [B,T]
{
    // Per-wave LDS: rows 0..17 = hidden[t0-1 .. t0+16] (zero-padded at edges),
    // rows 18..20 = conv taps w0,w1,w2, row 21 = conv bias.  33-col pad.
    __shared__ float tile[WAVES_PER_BLOCK][22][33];

    const int wave    = threadIdx.x >> 5;
    const int lane    = threadIdx.x & 31;
    const int tile_id = blockIdx.x * WAVES_PER_BLOCK + wave;
    const int b       = tile_id >> 7;          // / (T/16)
    const int t0      = (tile_id & 127) << 4;  // * 16
    const int g       = lane >> 4;             // lane half-group
    const int m       = lane & 15;             // A-matrix row (time within tile)
    const float bias  = lin_b[0];

    v8f acc = {};   // 16x16 f32 accumulator; all columns identical (B broadcast)

    for (int chunk = 0; chunk < CIF_H / 32; ++chunk) {
        const int hbase = chunk * 32;

        __syncthreads();   // previous iteration's LDS reads done
        // Coalesced cooperative load: lane -> channel hbase+lane, 18 time rows.
        const int hc = hbase + lane;
        for (int r = 0; r < 18; ++r) {
            const int t = t0 - 1 + r;
            float v = 0.0f;
            if (t >= 0 && t < CIF_T)
                v = hidden[((long)b * CIF_T + t) * CIF_H + hc];
            tile[wave][r][lane] = v;
        }
        tile[wave][18][lane] = conv_w[3 * hc + 0];
        tile[wave][19][lane] = conv_w[3 * hc + 1];
        tile[wave][20][lane] = conv_w[3 * hc + 2];
        tile[wave][21][lane] = conv_b[hc];
        __syncthreads();

        // Build A (16x32 f16 activations) and B (32x16 f16 broadcast weights)
        // per the CDNA5 VGPR layouts (CDNA5 ISA §7.12.2).
        v16h a, bm;
#pragma unroll
        for (int j = 0; j < 16; ++j) {
            // A layout: lanes 0-15: K = j (j<8) / j+8 (j>=8); lanes 16-31: +8.
            const int ka = (j < 8) ? (j + 8 * g) : (j + 8 + 8 * g);
            const float hm1 = tile[wave][m + 0][ka];   // hidden[t-1]
            const float h0  = tile[wave][m + 1][ka];   // hidden[t]
            const float hp1 = tile[wave][m + 2][ka];   // hidden[t+1]
            const float w0  = tile[wave][18][ka];
            const float w1  = tile[wave][19][ka];
            const float w2  = tile[wave][20][ka];
            const float cb  = tile[wave][21][ka];
            // relu(hidden[t] + conv_b + depthwise_conv3(hidden)[t])
            float act = fmaf(w0, hm1, fmaf(w2, hp1, fmaf(w1, h0, h0 + cb)));
            act = fmaxf(act, 0.0f);
            a[j] = (_Float16)act;
            // B layout: same K order as A (K = j (j<8) / j+8 (j>=8); lanes 16-31: +8).
            // Broadcast along N, so every column of D = dot.
            bm[j] = (_Float16)lin_w[hbase + ka];
        }

        acc = __builtin_amdgcn_wmma_f32_16x16x32_f16(
            /*neg_a=*/false, a, /*neg_b=*/false, bm,
            /*c_mod=*/(short)0, acc, /*reuse_a=*/false, /*reuse_b=*/false);
    }

    // D layout: lane L, VGPR v -> row (v + 8*(L>>4)), col (L&15); cols equal.
    if (m == 0) {
#pragma unroll
        for (int v = 0; v < 8; ++v) {
            const int row = v + 8 * g;
            const float x = acc[v] + bias;
            alphas[(long)b * CIF_T + t0 + row] = 1.0f / (1.0f + __expf(-x));
        }
    }
}

// ---------------------------------------------------------------------------
// Kernel 2: integrate-and-fire scan fused with gather.  One block per batch;
// each thread owns 4 channels (float4).  The scalar (integrate, rank) state
// depends only on alphas, so every thread tracks it redundantly -> no barriers
// inside the time loop and the fire branch is uniform across the block.
// alphas for the whole batch row are staged in LDS (8 KB) so the dependent
// scalar chain never waits on VMEM; hidden rows are prefetched ahead.
// ---------------------------------------------------------------------------
__global__ __launch_bounds__(256) void cif_scan_kernel(
    const float* __restrict__ hidden,   // [B,T,H]
    const float* __restrict__ alphas,   // [B,T]
    float* __restrict__ out,            // cif_output | cif_length | fires
    int max_label_len)
{
    __shared__ float s_alpha[CIF_T];

    const int  b   = blockIdx.x;
    const int  tid = threadIdx.x;
    const int  h4  = tid * 4;
    const long hidb = (long)b * CIF_T * CIF_H;

    float* __restrict__ cif_out = out;                                       // [B,ML,H]
    float* __restrict__ cif_len = out + (long)CIF_B * max_label_len * CIF_H; // [B]
    float* __restrict__ fires   = cif_len + CIF_B;                           // [B,T]

    // Stage the batch row's alphas into LDS (coalesced, 8 per thread).
#pragma unroll
    for (int i = 0; i < CIF_T / 256; ++i) {
        const int t = i * 256 + tid;
        s_alpha[t] = alphas[b * CIF_T + t];
    }
    __syncthreads();

    float f0 = 0.f, f1 = 0.f, f2 = 0.f, f3 = 0.f;  // frame (4 channels)
    float integrate = 0.f;
    float suma = 0.f;
    int   rank = 0;

    for (int t = 0; t < CIF_T; ++t) {
        // Keep the hidden-row stream ahead of consumption (global_prefetch_b8).
        if (t + PREFETCH_AHEAD < CIF_T)
            __builtin_prefetch(hidden + hidb + (long)(t + PREFETCH_AHEAD) * CIF_H + h4,
                               0, 3);

        const float alpha = s_alpha[t];                // LDS, off the VMEM path
        const float dist  = 1.0f - integrate;          // distribution_completion
        integrate += alpha;
        const bool fire = (integrate >= 1.0f);         // THRESH
        if (tid == 0) {
            fires[(long)b * CIF_T + t] = integrate;    // fires_t (pre-subtract)
            suma += alpha;
        }
        const float cur = fire ? dist : alpha;
        const float rem = alpha - cur;

        const float4 h = *(const float4*)(hidden + hidb + (long)t * CIF_H + h4);
        f0 = fmaf(cur, h.x, f0);
        f1 = fmaf(cur, h.y, f1);
        f2 = fmaf(cur, h.z, f2);
        f3 = fmaf(cur, h.w, f3);

        if (fire) {                                    // uniform branch
            if (rank < max_label_len) {
                float4 fr; fr.x = f0; fr.y = f1; fr.z = f2; fr.w = f3;
                *(float4*)(cif_out + ((long)(b * max_label_len + rank)) * CIF_H + h4) = fr;
            }
            f0 = rem * h.x; f1 = rem * h.y; f2 = rem * h.z; f3 = rem * h.w;
            ++rank;
            integrate -= 1.0f;
        }
    }

    // Zero-fill unfired rows (d_out is poisoned with 0xAA by the harness).
    for (int r = rank; r < max_label_len; ++r) {
        float4 z; z.x = 0.f; z.y = 0.f; z.z = 0.f; z.w = 0.f;
        *(float4*)(cif_out + ((long)(b * max_label_len + r)) * CIF_H + h4) = z;
    }
    if (tid == 0) cif_len[b] = suma;
}

// ---------------------------------------------------------------------------
extern "C" void kernel_launch(void* const* d_in, const int* in_sizes, int n_in,
                              void* d_out, int out_size, void* d_ws, size_t ws_size,
                              hipStream_t stream) {
    (void)in_sizes; (void)n_in; (void)ws_size;
    const float* hidden = (const float*)d_in[0];
    const float* conv_w = (const float*)d_in[1];
    const float* conv_b = (const float*)d_in[2];
    const float* lin_w  = (const float*)d_in[3];
    const float* lin_b  = (const float*)d_in[4];
    float* out    = (float*)d_out;
    float* alphas = (float*)d_ws;   // B*T floats = 128 KB scratch

    // d_out = cif_output [B,ML,H] ++ cif_length [B] ++ fires [B,T]
    int ML = (out_size - CIF_B - CIF_B * CIF_T) / (CIF_B * CIF_H);
    if (ML < 0) ML = 0;

    const int tiles  = (CIF_B * CIF_T) / 16;            // 2048 wave-tiles
    const int blocks = tiles / WAVES_PER_BLOCK;         // 256 blocks x 256 thr
    cif_alphas_kernel<<<blocks, 256, 0, stream>>>(hidden, conv_w, conv_b,
                                                  lin_w, lin_b, alphas);
    cif_scan_kernel<<<CIF_B, 256, 0, stream>>>(hidden, alphas, out, ML);
}
